// TensorRingLayer_18356690223125
// MI455X (gfx1250) — compile-verified, matrix operand order checked by reading, not run on hardware
//
#include <hip/hip_runtime.h>
#include <stdint.h>
#include <stddef.h>

// ---------------------------------------------------------------------------
// Tensor-Ring layer for MI455X (gfx1250), wave32 + WMMA bf16.
//
//   out[b, o] = sum_{r,s} env[b,r,s] * sum_x G[r,o,x,s] * x[b,x]   (+bias)
//
// Main pass: one GEMM per core, (4096 x 32768) @ (32768 x 512), with the A
// operand built on the fly as x[b,x]*env[b,r,s] (per 32-wide K chunk the
// (r,s) scale has fixed r and varies with s).  K ordered (r, x-block, x, s) so every
// B-tile row is 32 contiguous floats (128B vector loads); each 64 MiB core is
// L2-resident while its blocks run (blockIdx.z = core is slowest).
// bf16 WMMA 16x16x32 with f32 accumulation; chain math stays f32.
// Inner loop is software-pipelined: global loads for chunk k+1 issue first,
// then fragment ds_loads + 4 back-to-back WMMAs for chunk k, then the
// convert+ds_store half of staging (hidden under the matrix pipe).
// Workspace use: ~16.5 MiB of d_ws.
// ---------------------------------------------------------------------------

typedef __attribute__((ext_vector_type(16))) __bf16 v16bf;
typedef __attribute__((ext_vector_type(8)))  __bf16 v8bf;
typedef __attribute__((ext_vector_type(8)))  float  v8f;

#define B_TOTAL   4096
#define DIN       2048
#define DOUT      2048
#define LSEG      512
#define CORE_R_STRIDE (512*512*8)   // 2097152 elements between r planes
#define CORE_O_STRIDE (512*8)       // 4096 elements between o rows

// workspace layout (in floats)
static constexpr size_t WS_PART = 0;                        // 4*8*16*4096 = 2097152
static constexpr size_t WS_GSUM = WS_PART + 2097152;        // 4*8*4096    =  131072
static constexpr size_t WS_S    = WS_GSUM + 131072;         // 4096*256    = 1048576
static constexpr size_t WS_ENV  = WS_S    + 1048576;        // 4096*256    = 1048576

// ---------------------------------------------------------------------------
// K1a: partial reduction of cores over o.  g_part[c][r][oc][x*8+s] =
//      sum_{o in oc-chunk} core[r,o,x,s].  512 blocks x 256 threads,
//      fully coalesced (inner (x,s) plane of 4096 floats is contiguous).
// ---------------------------------------------------------------------------
__global__ __launch_bounds__(256) void k_core_reduce_partial(
    const float* __restrict__ c0, const float* __restrict__ c1,
    const float* __restrict__ c2, const float* __restrict__ c3,
    float* __restrict__ ws) {
  const int blk = blockIdx.x;           // 512
  const int c  = blk >> 7;
  const int r  = (blk >> 4) & 7;
  const int oc = blk & 15;
  const float* cp = (c == 0) ? c0 : (c == 1) ? c1 : (c == 2) ? c2 : c3;
  const float* base = cp + (size_t)r * CORE_R_STRIDE + (size_t)(oc * 32) * CORE_O_STRIDE;
  const int t = threadIdx.x;
  float acc[16];
#pragma unroll
  for (int e = 0; e < 16; ++e) acc[e] = 0.f;
  for (int o = 0; o < 32; ++o) {
    const float* bo = base + (size_t)o * CORE_O_STRIDE;
#pragma unroll
    for (int e = 0; e < 16; ++e) acc[e] += bo[e * 256 + t];
  }
  float* dst = ws + WS_PART + ((size_t)((c * 8 + r) * 16 + oc)) * 4096;
#pragma unroll
  for (int e = 0; e < 16; ++e) dst[e * 256 + t] = acc[e];
}

// ---------------------------------------------------------------------------
// K1b: g_sum[c][r][x*8+s] = sum over 16 o-chunks (deterministic, no atomics)
// ---------------------------------------------------------------------------
__global__ __launch_bounds__(256) void k_core_reduce_final(float* __restrict__ ws) {
  const size_t gid = (size_t)blockIdx.x * 256 + threadIdx.x;  // 131072 total
  const size_t cr  = gid >> 12;
  const size_t idx = gid & 4095;
  const float* src = ws + WS_PART + cr * 16 * 4096 + idx;
  float a = 0.f;
#pragma unroll
  for (int ch = 0; ch < 16; ++ch) a += src[(size_t)ch * 4096];
  ws[WS_GSUM + gid] = a;
}

// ---------------------------------------------------------------------------
// K2: S[b][c][r][s] = sum_x x[b, c*512+x] * g_sum[c][r][x][s]   (f32)
// ---------------------------------------------------------------------------
__global__ __launch_bounds__(256) void k_compute_S(
    const float* __restrict__ x, const float* __restrict__ gsum,
    float* __restrict__ Sout) {
  __shared__ float xrow[DIN];
  const int b = blockIdx.x;
  const int t = threadIdx.x;
  for (int i = t; i < DIN; i += 256) xrow[i] = x[(size_t)b * DIN + i];
  __syncthreads();
  const int c = t >> 6, r = (t >> 3) & 7, s = t & 7;
  const float* gp = gsum + (size_t)c * 32768 + (size_t)r * 4096 + s;
  const float* xp = xrow + c * LSEG;
  float acc = 0.f;
#pragma unroll 8
  for (int xx = 0; xx < LSEG; ++xx) acc += xp[xx] * gp[(size_t)xx * 8];
  Sout[(size_t)b * 256 + t] = acc;   // t == c*64 + r*8 + s
}

// ---------------------------------------------------------------------------
// K3: per batch row, chain the 8x8 transfer matrices and form env_i =
//     (Q_i @ P_i)^T.  64 threads = one 8x8 element per thread.
// ---------------------------------------------------------------------------
__global__ __launch_bounds__(64) void k_compute_env(
    const float* __restrict__ S, float* __restrict__ env) {
  __shared__ float Sm[4][64];
  __shared__ float Pm[4][64];
  __shared__ float Qm[4][64];
  const int b = blockIdx.x;
  const int t = threadIdx.x;
  const int r  = t >> 3;
  const int cc = t & 7;
#pragma unroll
  for (int i = 0; i < 4; ++i) Sm[i][t] = S[(size_t)b * 256 + i * 64 + t];
  __syncthreads();
  Pm[0][t] = (r == cc) ? 1.f : 0.f;   // P0 = I
  Pm[1][t] = Sm[0][t];                // P1 = S0
  Qm[3][t] = (r == cc) ? 1.f : 0.f;   // Q3 = I
  Qm[2][t] = Sm[3][t];                // Q2 = S3
  __syncthreads();
  {
    float a = 0.f, q = 0.f;
#pragma unroll
    for (int k = 0; k < 8; ++k) {
      a += Pm[1][r * 8 + k] * Sm[1][k * 8 + cc];   // P2 = P1 @ S1
      q += Sm[2][r * 8 + k] * Qm[2][k * 8 + cc];   // Q1 = S2 @ Q2
    }
    Pm[2][t] = a;
    Qm[1][t] = q;
  }
  __syncthreads();
  {
    float a = 0.f, q = 0.f;
#pragma unroll
    for (int k = 0; k < 8; ++k) {
      a += Pm[2][r * 8 + k] * Sm[2][k * 8 + cc];   // P3 = P2 @ S2
      q += Sm[1][r * 8 + k] * Qm[1][k * 8 + cc];   // Q0 = S1 @ Q1
    }
    Pm[3][t] = a;
    Qm[0][t] = q;
  }
  __syncthreads();
#pragma unroll
  for (int i = 0; i < 4; ++i) {       // env_i[r,s] = sum_p Q_i[s,p] P_i[p,r]
    float v = 0.f;
#pragma unroll
    for (int p = 0; p < 8; ++p) v += Qm[i][cc * 8 + p] * Pm[i][p * 8 + r];
    env[(size_t)b * 256 + i * 64 + t] = v;
  }
}

// ---------------------------------------------------------------------------
// K4: main GEMM.  Block tile 64(b) x 128(o), 8 wave32s in a 4x2 wave grid
// (each wave: 16b x 64o = one A-frag + 4 B-frags + 4 f32 accumulators).
// ---------------------------------------------------------------------------
__global__ __launch_bounds__(256) void k_ring_main(
    const float* __restrict__ x,
    const float* __restrict__ c0, const float* __restrict__ c1,
    const float* __restrict__ c2, const float* __restrict__ c3,
    const float* __restrict__ bias,
    const float* __restrict__ env,
    float* __restrict__ out) {
  __shared__ __align__(16) __bf16 at[2][64][40];    // A tile, padded rows
  __shared__ __align__(16) __bf16 bt[2][128][40];   // B tile, padded rows
  __shared__ float env_s[64][65];                   // padded vs bank conflicts

  const int tid = threadIdx.x;
  const int bx = blockIdx.x;   // o-tile   (0..3)
  const int by = blockIdx.y;   // b-tile   (0..63)
  const int cz = blockIdx.z;   // core     (0..3)
  const float* cp = (cz == 0) ? c0 : (cz == 1) ? c1 : (cz == 2) ? c2 : c3;
  const float* gB0 = cp + (size_t)(bx * 128) * CORE_O_STRIDE;

  // env for this (b-tile, core) into LDS
  for (int idx = tid; idx < 64 * 64; idx += 256)
    env_s[idx >> 6][idx & 63] =
        env[((size_t)(by * 64 + (idx >> 6)) * 4 + cz) * 64 + (idx & 63)];

  const int lane = tid & 31;
  const int wave = tid >> 5;
  const int wb = (wave & 3) * 16;    // wave's b offset in tile
  const int wo = (wave >> 2) * 64;   // wave's o offset in tile
  const int mrow = wb + (lane & 15);
  const int klo  = (lane & 16) ? 8 : 0;   // ISA 16-bit operand lane striping
  const int nrow = wo + (lane & 15);      // base B-frag row; +a*16 per frag

  v8f acc[4];
#pragma unroll
  for (int a = 0; a < 4; ++a) acc[a] = (v8f){0.f,0.f,0.f,0.f,0.f,0.f,0.f,0.f};

  // staging thread mapping
  const int brow  = tid >> 2;   // A: b row  (0..63), 4 threads per row
  const int bg    = tid & 3;    // A: x group (4 x values per chunk)
  const int orow  = tid >> 1;   // B: o row  (0..127), 2 threads per row
  const int ohalf = tid & 1;
  const float* xbase = x + (size_t)(by * 64 + brow) * DIN + cz * LSEG;
  const float* gBrow = gB0 + (size_t)orow * CORE_O_STRIDE + ohalf * 16;

  __syncthreads();   // env_s visible before first staging

  // ---- prologue: stage chunk 0 into buffer 0 directly
  {
    const float4 g0 = *(const float4*)(gBrow + 0);
    const float4 g1 = *(const float4*)(gBrow + 4);
    const float4 g2 = *(const float4*)(gBrow + 8);
    const float4 g3 = *(const float4*)(gBrow + 12);
    v8bf bv0, bv1;
    bv0[0]=(__bf16)g0.x; bv0[1]=(__bf16)g0.y; bv0[2]=(__bf16)g0.z; bv0[3]=(__bf16)g0.w;
    bv0[4]=(__bf16)g1.x; bv0[5]=(__bf16)g1.y; bv0[6]=(__bf16)g1.z; bv0[7]=(__bf16)g1.w;
    bv1[0]=(__bf16)g2.x; bv1[1]=(__bf16)g2.y; bv1[2]=(__bf16)g2.z; bv1[3]=(__bf16)g2.w;
    bv1[4]=(__bf16)g3.x; bv1[5]=(__bf16)g3.y; bv1[6]=(__bf16)g3.z; bv1[7]=(__bf16)g3.w;
    *(v8bf*)(&bt[0][orow][ohalf * 16])     = bv0;
    *(v8bf*)(&bt[0][orow][ohalf * 16 + 8]) = bv1;
    const float xv = xbase[bg];
    v8bf av;
#pragma unroll
    for (int s = 0; s < 8; ++s) av[s] = (__bf16)(xv * env_s[brow][s]);
    *(v8bf*)(&at[0][brow][bg * 8]) = av;
  }

#pragma unroll 1
  for (int kc = 0; kc < 1024; ++kc) {
    const int p = kc & 1;
    const bool has_next = (kc + 1) < 1024;        // uniform
    const int kn = has_next ? (kc + 1) : kc;
    const int rn = kn >> 7;
    const int xn = kn & 127;

    __syncthreads();   // buffer p staged; everyone done reading p^1

    // ---- (1) issue next-chunk global loads first (pure latency)
    float4 g0{}, g1{}, g2{}, g3{};
    float xv = 0.f;
    float ev[8];
#pragma unroll
    for (int s = 0; s < 8; ++s) ev[s] = 0.f;
    if (has_next) {
      const float* gB = gBrow + (size_t)rn * CORE_R_STRIDE + (size_t)xn * 32;
      __builtin_prefetch(gB + 32, 0, 1);          // next x-block, same row
      g0 = *(const float4*)(gB + 0);
      g1 = *(const float4*)(gB + 4);
      g2 = *(const float4*)(gB + 8);
      g3 = *(const float4*)(gB + 12);
      xv = xbase[xn * 4 + bg];
#pragma unroll
      for (int s = 0; s < 8; ++s) ev[s] = env_s[brow][rn * 8 + s];
    }

    // ---- (2) current-chunk fragments, then 4 back-to-back WMMAs
    const v8bf alo = *(const v8bf*)(&at[p][mrow][klo]);
    const v8bf ahi = *(const v8bf*)(&at[p][mrow][klo + 16]);
    const v16bf afrag = __builtin_shufflevector(
        alo, ahi, 0,1,2,3,4,5,6,7,8,9,10,11,12,13,14,15);
    v16bf bfr[4];
#pragma unroll
    for (int a = 0; a < 4; ++a) {
      const v8bf blo = *(const v8bf*)(&bt[p][nrow + a * 16][klo]);
      const v8bf bhi = *(const v8bf*)(&bt[p][nrow + a * 16][klo + 16]);
      bfr[a] = __builtin_shufflevector(
          blo, bhi, 0,1,2,3,4,5,6,7,8,9,10,11,12,13,14,15);
    }
#pragma unroll
    for (int a = 0; a < 4; ++a)
      acc[a] = __builtin_amdgcn_wmma_f32_16x16x32_bf16(
          false, afrag, false, bfr[a], (short)0, acc[a], false, false);

    // ---- (3) convert + store next chunk (hidden under the matrix pipe)
    if (has_next) {
      const int pn = p ^ 1;
      v8bf bv0, bv1;
      bv0[0]=(__bf16)g0.x; bv0[1]=(__bf16)g0.y; bv0[2]=(__bf16)g0.z; bv0[3]=(__bf16)g0.w;
      bv0[4]=(__bf16)g1.x; bv0[5]=(__bf16)g1.y; bv0[6]=(__bf16)g1.z; bv0[7]=(__bf16)g1.w;
      bv1[0]=(__bf16)g2.x; bv1[1]=(__bf16)g2.y; bv1[2]=(__bf16)g2.z; bv1[3]=(__bf16)g2.w;
      bv1[4]=(__bf16)g3.x; bv1[5]=(__bf16)g3.y; bv1[6]=(__bf16)g3.z; bv1[7]=(__bf16)g3.w;
      *(v8bf*)(&bt[pn][orow][ohalf * 16])     = bv0;
      *(v8bf*)(&bt[pn][orow][ohalf * 16 + 8]) = bv1;
      v8bf av;
#pragma unroll
      for (int s = 0; s < 8; ++s) av[s] = (__bf16)(xv * ev[s]);
      *(v8bf*)(&at[pn][brow][bg * 8]) = av;
    }
  }

  // epilogue: C layout is VGPR j -> M=j (lanes 0-15) / M=j+8 (lanes 16-31)
#pragma unroll
  for (int a = 0; a < 4; ++a) {
    const int col = cz * 512 + bx * 128 + wo + a * 16 + (lane & 15);
    const float bv = bias[col];
#pragma unroll
    for (int v = 0; v < 8; ++v) {
      const int row = by * 64 + wb + v + ((lane & 16) >> 1);
      out[(size_t)row * DOUT + col] = acc[a][v] + bv;
    }
  }
}

// ---------------------------------------------------------------------------
extern "C" void kernel_launch(void* const* d_in, const int* in_sizes, int n_in,
                              void* d_out, int out_size, void* d_ws, size_t ws_size,
                              hipStream_t stream) {
  (void)in_sizes; (void)n_in; (void)out_size; (void)ws_size;
  const float* x    = (const float*)d_in[0];
  const float* c0   = (const float*)d_in[1];
  const float* c1   = (const float*)d_in[2];
  const float* c2   = (const float*)d_in[3];
  const float* c3   = (const float*)d_in[4];
  const float* bias = (const float*)d_in[5];
  float* out = (float*)d_out;
  float* ws  = (float*)d_ws;   // needs ~16.5 MiB

  k_core_reduce_partial<<<512, 256, 0, stream>>>(c0, c1, c2, c3, ws);
  k_core_reduce_final<<<512, 256, 0, stream>>>(ws);
  k_compute_S<<<B_TOTAL, 256, 0, stream>>>(x, ws + WS_GSUM, ws + WS_S);
  k_compute_env<<<B_TOTAL, 64, 0, stream>>>(ws + WS_S, ws + WS_ENV);
  dim3 grid(4, 64, 4);   // (o-tiles, b-tiles, core) — core slowest for L2 reuse
  k_ring_main<<<grid, 256, 0, stream>>>(x, c0, c1, c2, c3, bias,
                                        ws + WS_ENV, out);
}
